// Quant_res_pw2_84799834293032
// MI455X (gfx1250) — matrix operand order repaired and checked by reading, not compiled, not run on hardware
//
#include <hip/hip_runtime.h>

// ---------------------------------------------------------------------------
// CDNA5 / gfx1250 implementation of the quantized depthwise-conv pipeline.
// Dominant matmuls (VQ nearest-code searches) run on v_wmma_f32_16x16x32_bf16,
// K=64 double-buffered LDS staging (1 barrier / chunk), M=64 blocking in the
// dominant fused conv+VQ kernel for 4x B-operand reuse out of L2.
// ---------------------------------------------------------------------------

typedef __attribute__((ext_vector_type(16))) __bf16 v16bf;
typedef __attribute__((ext_vector_type(8)))  float  v8f;

#define HW   3136   // 56*56
#define IMGW 56
#define BATCH 4
#define OUPC 16
#define NCH  49     // 3136 / 64 K-chunks

struct Raw32 { uint4 a, b; };   // 32 bytes == v16bf

static __device__ __forceinline__ unsigned short f32_to_bf16(float f) {
  unsigned int u = __builtin_bit_cast(unsigned int, f);
  u += 0x7fffu + ((u >> 16) & 1u);          // round to nearest even
  return (unsigned short)(u >> 16);
}

// Load 16 contiguous bf16 values (32B, 16B-aligned) into a v16bf fragment.
static __device__ __forceinline__ v16bf load_v16bf(const unsigned short* p) {
  const uint4* p4 = reinterpret_cast<const uint4*>(p);
  Raw32 r; r.a = p4[0]; r.b = p4[1];
  return __builtin_bit_cast(v16bf, r);
}

// A-fragment for 16-bit WMMA: row = lane&15; lanes<16 take K{0..15},
// lanes>=16 take K{16..31}.  rowp = 32 bf16 values of that A row.
static __device__ __forceinline__ v16bf load_a_frag(const unsigned short* rowp, int lane) {
  const uint4* p4 = reinterpret_cast<const uint4*>(rowp);
  const int c = (lane < 16) ? 0 : 1;        // 16-element half index
  Raw32 r; r.a = p4[2 * c]; r.b = p4[2 * c + 1];
  return __builtin_bit_cast(v16bf, r);
}

static __device__ __forceinline__ v8f wmma_bf16(v16bf a, v16bf b, v8f c) {
  return __builtin_amdgcn_wmma_f32_16x16x32_bf16(false, a, false, b,
                                                 (short)0, c, false, false);
}

// ---------------------------------------------------------------------------
// Prep: f32 codebook -> bf16 copy + 0.5*|c|^2 per row.
// ---------------------------------------------------------------------------
__global__ __launch_bounds__(256)
void k_prep_codes(const float* __restrict__ src, unsigned short* __restrict__ dst,
                  float* __restrict__ hn, int dim) {
  __shared__ float red[256];
  const int row = blockIdx.x, tid = threadIdx.x;
  const float* s = src + (size_t)row * dim;
  unsigned short* d = dst + (size_t)row * dim;
  float ss = 0.f;
  for (int i = tid; i < dim; i += 256) { float v = s[i]; d[i] = f32_to_bf16(v); ss += v * v; }
  red[tid] = ss; __syncthreads();
  for (int o = 128; o > 0; o >>= 1) { if (tid < o) red[tid] += red[tid + o]; __syncthreads(); }
  if (tid == 0) hn[row] = 0.5f * red[0];
}

// ---------------------------------------------------------------------------
// Tiny VQ for 3x3 filters (dim 9, 256 codes), exact f32. One thread per filter.
// ---------------------------------------------------------------------------
__global__ __launch_bounds__(256)
void k_vq_w(const float* __restrict__ w, const float* __restrict__ emb,
            float* __restrict__ outw) {
  const int r = threadIdx.x;
  float wv[9];
#pragma unroll
  for (int t = 0; t < 9; ++t) wv[t] = w[r * 9 + t];
  float best = 3.4e38f; int bi = 0;
  for (int k = 0; k < 256; ++k) {
    float d2 = 0.f;
#pragma unroll
    for (int t = 0; t < 9; ++t) { float df = wv[t] - emb[k * 9 + t]; d2 = fmaf(df, df, d2); }
    if (d2 < best) { best = d2; bi = k; }   // first-occurrence tie break
  }
#pragma unroll
  for (int t = 0; t < 9; ++t) outw[r * 9 + t] = emb[bi * 9 + t];
}

// ---------------------------------------------------------------------------
// WMMA nearest-code search: A (R x 3136) f32 vs 512 bf16 codes -> idx[R].
// Block = 16 rows, 512 threads = 16 waves; wave w covers codes {grp*256+w*16..}.
// K=64 per chunk, double-buffered A tile, 4 WMMAs per barrier.
// argmax of (dot - 0.5|c|^2)  ==  argmin of squared distance.
// ---------------------------------------------------------------------------
__global__ __launch_bounds__(512)
void k_vq_rows(const float* __restrict__ A, const unsigned short* __restrict__ codes,
               const float* __restrict__ hn, int* __restrict__ idx_out) {
  __shared__ __align__(16) unsigned short atile[2][16 * 64];
  __shared__ float scores[16 * 512];
  __shared__ float pbv[16 * 32];
  __shared__ int   pbi[16 * 32];
  const int tid = threadIdx.x, wave = tid >> 5, lane = tid & 31;
  const int row0 = blockIdx.x * 16;
  const float* arow = A + (size_t)(row0 + wave) * HW;   // this thread stages row `wave`

  auto stage = [&](int buf, int kc) {
    const float2 v = *reinterpret_cast<const float2*>(arow + kc * 64 + 2 * lane);
    const unsigned int packed = (unsigned int)f32_to_bf16(v.x) |
                                ((unsigned int)f32_to_bf16(v.y) << 16);
    *reinterpret_cast<unsigned int*>(&atile[buf][wave * 64 + 2 * lane]) = packed;
  };

  const int n = lane & 15;
  const unsigned short* c0 = codes + (size_t)((wave << 4) + n) * HW + ((lane < 16) ? 0 : 16);
  const unsigned short* c1 = c0 + (size_t)256 * HW;

  v8f acc0 = {}, acc1 = {};
  stage(0, 0);
  __syncthreads();
  for (int kc = 0; kc < NCH; ++kc) {
    const int cur = kc & 1;
    if (kc + 1 < NCH) stage(cur ^ 1, kc + 1);
    const unsigned short* ar = &atile[cur][(lane & 15) * 64];
    const v16bf a0 = load_a_frag(ar, lane);
    const v16bf a1 = load_a_frag(ar + 32, lane);
    const int kb = kc * 64;
    acc0 = wmma_bf16(a0, load_v16bf(c0 + kb), acc0);
    acc0 = wmma_bf16(a1, load_v16bf(c0 + kb + 32), acc0);
    acc1 = wmma_bf16(a0, load_v16bf(c1 + kb), acc1);
    acc1 = wmma_bf16(a1, load_v16bf(c1 + kb + 32), acc1);
    __syncthreads();
  }

  // scores[g][code] = dot - 0.5|c|^2 for all 512 codes
  {
    const int codeL = (wave << 4) + n;
    const float h0 = hn[codeL], h1 = hn[256 + codeL];
#pragma unroll
    for (int r = 0; r < 8; ++r) {
      const int m = (lane < 16) ? r : r + 8;
      scores[m * 512 + codeL]       = acc0[r] - h0;
      scores[m * 512 + 256 + codeL] = acc1[r] - h1;
    }
  }
  __syncthreads();
  // parallel partial argmax: thread (g=wave, seg=lane) scans 16 codes
  {
    const float* srow = scores + wave * 512 + lane * 16;
    float bv = srow[0]; int bi = 0;
#pragma unroll
    for (int k = 1; k < 16; ++k) { const float v = srow[k]; if (v > bv) { bv = v; bi = k; } }
    pbv[wave * 32 + lane] = bv; pbi[wave * 32 + lane] = lane * 16 + bi;
  }
  __syncthreads();
  if (tid < 16) {     // first-occurrence tie-break: ascending seg + strict >
    const float* pv = pbv + tid * 32; const int* pi = pbi + tid * 32;
    float bv = pv[0]; int bi = pi[0];
    for (int s = 1; s < 32; ++s) { const float v = pv[s]; if (v > bv) { bv = v; bi = pi[s]; } }
    idx_out[row0 + tid] = bi;
  }
}

// ---------------------------------------------------------------------------
// Gather quantized rows: out[r] = codes[idx[r]]
// ---------------------------------------------------------------------------
__global__ __launch_bounds__(256)
void k_gather(const int* __restrict__ idx, const float* __restrict__ codes,
              float* __restrict__ outr, int dim) {
  const int row = blockIdx.x; const int c = idx[row];
  const float* s = codes + (size_t)c * dim;
  float* d = outr + (size_t)row * dim;
  for (int i = threadIdx.x; i < dim; i += 256) d[i] = s[i];
}

// ---------------------------------------------------------------------------
// Depthwise conv 1: out(b,ch) = conv3x3(in(b, ch%16), w[ch]); 1024 blocks.
// ---------------------------------------------------------------------------
__global__ __launch_bounds__(256)
void k_dwconv1(const float* __restrict__ in, const float* __restrict__ wq,
               float* __restrict__ outc) {
  __shared__ float img[HW];
  const int blk = blockIdx.x;
  const int b = blk >> 8, ch = blk & 255;
  const float* s = in + (size_t)(b * 16 + (ch & 15)) * HW;
  for (int i = threadIdx.x; i < HW; i += 256) img[i] = s[i];
  float f[9];
#pragma unroll
  for (int t = 0; t < 9; ++t) f[t] = wq[ch * 9 + t];
  __syncthreads();
  for (int pix = threadIdx.x; pix < HW; pix += 256) {
    const int y = pix / IMGW, x = pix % IMGW;
    float acc = 0.f;
#pragma unroll
    for (int dy = 0; dy < 3; ++dy) {
      const int yy = y + dy - 1;
#pragma unroll
      for (int dx = 0; dx < 3; ++dx) {
        const int xx = x + dx - 1;
        if (yy >= 0 && yy < IMGW && xx >= 0 && xx < IMGW)
          acc = fmaf(img[yy * IMGW + xx], f[dy * 3 + dx], acc);
      }
    }
    outc[(size_t)blk * HW + pix] = acc;
  }
}

// ---------------------------------------------------------------------------
// Fused conv2 + VQ(emb_end) + code counting, M=64 blocked.
// Block = (b, c2 group of 4); 512 threads = 16 waves. A-tile rows: mt*16+g
// where mt indexes the 4 consecutive c2 values (all share the same filter
// group c2base>>4, so each thread needs a single 9-tap filter for g = wave).
// Each wave owns one 16-code N-tile and 4 M-tiles -> every B fragment feeds
// 4 WMMAs (4x L2 traffic reduction vs M=16).
// ---------------------------------------------------------------------------
__global__ __launch_bounds__(512)
void k_conv2_vqend(const float* __restrict__ h3, const float* __restrict__ w2q,
                   const unsigned short* __restrict__ codes, const float* __restrict__ hn,
                   int* __restrict__ counts) {
  __shared__ float img[4][HW];
  __shared__ __align__(16) unsigned short atile[2][64 * 64];
  __shared__ float scores[16 * 256];
  __shared__ float pbv[16 * 16];
  __shared__ int   pbi[16 * 16];
  const int tid = threadIdx.x, wave = tid >> 5, lane = tid & 31;
  const int b = blockIdx.x >> 6, c2g = blockIdx.x & 63;
  const int c2base = c2g * 4;

  // load the 4 input images for this c2 group
  for (int q = 0; q < 4; ++q) {
    const float* src = h3 + (size_t)(b * 256 + c2base + q) * HW;
    for (int i = tid; i < HW; i += 512) img[q][i] = src[i];
  }
  // 9-tap filter for this thread's g = wave (same for all 4 c2 in the group)
  const float* fp = w2q + (wave * 16 + (c2base >> 4)) * 9;
  float f[9];
#pragma unroll
  for (int t = 0; t < 9; ++t) f[t] = fp[t];
  __syncthreads();

  auto conv_at = [&](int q, int pix) -> float {
    const int y = pix / IMGW, x = pix % IMGW;
    float cv = 0.f;
#pragma unroll
    for (int dy = 0; dy < 3; ++dy) {
      const int yy = y + dy - 1;
#pragma unroll
      for (int dx = 0; dx < 3; ++dx) {
        const int xx = x + dx - 1;
        if (yy >= 0 && yy < IMGW && xx >= 0 && xx < IMGW)
          cv = fmaf(img[q][yy * IMGW + xx], f[dy * 3 + dx], cv);
      }
    }
    return cv;
  };

  auto stage = [&](int buf, int kc) {
    const int pix = kc * 64 + 2 * lane;
#pragma unroll
    for (int mt = 0; mt < 4; ++mt) {
      const float v0 = conv_at(mt, pix);
      const float v1 = conv_at(mt, pix + 1);
      const unsigned int packed = (unsigned int)f32_to_bf16(v0) |
                                  ((unsigned int)f32_to_bf16(v1) << 16);
      *reinterpret_cast<unsigned int*>(&atile[buf][(mt * 16 + wave) * 64 + 2 * lane]) = packed;
    }
  };

  const int n = lane & 15;
  const unsigned short* cb = codes + (size_t)((wave << 4) + n) * HW + ((lane < 16) ? 0 : 16);

  v8f acc[4] = {{}, {}, {}, {}};
  stage(0, 0);
  __syncthreads();
  for (int kc = 0; kc < NCH; ++kc) {
    const int cur = kc & 1;
    if (kc + 1 < NCH) stage(cur ^ 1, kc + 1);
    const int kb = kc * 64;
    const v16bf b0 = load_v16bf(cb + kb);
    const v16bf b1 = load_v16bf(cb + kb + 32);
#pragma unroll
    for (int mt = 0; mt < 4; ++mt) {
      const unsigned short* ar = &atile[cur][(mt * 16 + (lane & 15)) * 64];
      acc[mt] = wmma_bf16(load_a_frag(ar, lane), b0, acc[mt]);
      acc[mt] = wmma_bf16(load_a_frag(ar + 32, lane), b1, acc[mt]);
    }
    __syncthreads();
  }

  // epilogue per M-tile: scores -> parallel argmax -> count
#pragma unroll
  for (int mt = 0; mt < 4; ++mt) {
    {
      const int code = (wave << 4) + n;
      const float h = hn[code];
#pragma unroll
      for (int r = 0; r < 8; ++r) {
        const int gm = (lane < 16) ? r : r + 8;
        scores[gm * 256 + code] = acc[mt][r] - h;
      }
    }
    __syncthreads();
    if (tid < 256) {            // thread (g = tid>>4, seg = tid&15) scans 16 codes
      const int g = tid >> 4, seg = tid & 15;
      const float* srow = scores + g * 256 + seg * 16;
      float bv = srow[0]; int bi = 0;
#pragma unroll
      for (int k = 1; k < 16; ++k) { const float v = srow[k]; if (v > bv) { bv = v; bi = k; } }
      pbv[g * 16 + seg] = bv; pbi[g * 16 + seg] = seg * 16 + bi;
    }
    __syncthreads();
    if (tid < 16) {             // g = tid; first-occurrence tie-break preserved
      const float* pv = pbv + tid * 16; const int* pi = pbi + tid * 16;
      float bv = pv[0]; int bi = pi[0];
      for (int s = 1; s < 16; ++s) { const float v = pv[s]; if (v > bv) { bv = v; bi = pi[s]; } }
      atomicAdd(&counts[(b * OUPC + tid) * 256 + bi], 1);
    }
    __syncthreads();
  }
}

// ---------------------------------------------------------------------------
// out_pre[b,g,:] = counts[b,g,:] @ emb_end   (64 x 3136 outputs)
// ---------------------------------------------------------------------------
__global__ __launch_bounds__(256)
void k_outpre(const float* __restrict__ emb_end, const int* __restrict__ counts,
              float* __restrict__ outpre) {
  __shared__ float cw[256];
  const int bg = blockIdx.x, tid = threadIdx.x;
  cw[tid] = (float)counts[bg * 256 + tid];
  __syncthreads();
  for (int pix = tid; pix < HW; pix += 256) {
    float acc = 0.f;
    for (int k = 0; k < 256; ++k) acc = fmaf(cw[k], emb_end[(size_t)k * HW + pix], acc);
    outpre[(size_t)bg * HW + pix] = acc;
  }
}

// ---------------------------------------------------------------------------
// BN train-mode statistics per channel g over (B, H, W). Deterministic reduce.
// ---------------------------------------------------------------------------
__global__ __launch_bounds__(256)
void k_bnstats(const float* __restrict__ outpre, float* __restrict__ stats) {
  __shared__ float rs[256], rss[256];
  const int g = blockIdx.x, tid = threadIdx.x;
  float s = 0.f, ss = 0.f;
  for (int b = 0; b < BATCH; ++b)
    for (int pix = tid; pix < HW; pix += 256) {
      const float v = outpre[(size_t)(b * OUPC + g) * HW + pix];
      s += v; ss = fmaf(v, v, ss);
    }
  rs[tid] = s; rss[tid] = ss; __syncthreads();
  for (int o = 128; o > 0; o >>= 1) {
    if (tid < o) { rs[tid] += rs[tid + o]; rss[tid] += rss[tid + o]; }
    __syncthreads();
  }
  if (tid == 0) { stats[g] = rs[0]; stats[OUPC + g] = rss[0]; }
}

// ---------------------------------------------------------------------------
// Final: normalize, scale/shift, residual, ReLU.
// ---------------------------------------------------------------------------
__global__ __launch_bounds__(256)
void k_final(const float* __restrict__ outpre, const float* __restrict__ x,
             const float* __restrict__ stats, const float* __restrict__ gamma,
             const float* __restrict__ beta, float* __restrict__ out) {
  const int i = blockIdx.x * 256 + threadIdx.x;
  if (i >= BATCH * OUPC * HW) return;
  const int g = (i / HW) & (OUPC - 1);
  const float N = (float)(BATCH * HW);
  const float mean = stats[g] / N;
  const float var = stats[OUPC + g] / N - mean * mean;
  float v = (outpre[i] - mean) * rsqrtf(var + 1e-5f);
  v = fmaf(v, gamma[g], beta[g]) + x[i];
  out[i] = v > 0.f ? v : 0.f;
}

__global__ __launch_bounds__(256)
void k_zero_counts(int* __restrict__ counts) {
  const int i = blockIdx.x * 256 + threadIdx.x;
  if (i < BATCH * OUPC * 256) counts[i] = 0;
}

// ---------------------------------------------------------------------------
extern "C" void kernel_launch(void* const* d_in, const int* in_sizes, int n_in,
                              void* d_out, int out_size, void* d_ws, size_t ws_size,
                              hipStream_t stream) {
  const float* x       = (const float*)d_in[0];
  const float* pw_w1   = (const float*)d_in[1];
  const float* pw_w2   = (const float*)d_in[2];
  const float* emb_f1  = (const float*)d_in[3];
  const float* emb_pw1 = (const float*)d_in[4];
  const float* emb_f2  = (const float*)d_in[5];
  const float* emb_pw2 = (const float*)d_in[6];
  const float* emb_end = (const float*)d_in[7];
  const float* gamma   = (const float*)d_in[8];
  const float* beta    = (const float*)d_in[9];
  float* out = (float*)d_out;

  char* ws = (char*)d_ws;
  size_t off = 0;
  auto alloc = [&](size_t bytes) -> char* {
    char* p = ws + off; off += (bytes + 255) & ~(size_t)255; return p;
  };
  unsigned short* ef1b  = (unsigned short*)alloc((size_t)512 * HW * 2);
  unsigned short* ef2b  = (unsigned short*)alloc((size_t)512 * HW * 2);
  unsigned short* eendb = (unsigned short*)alloc((size_t)256 * HW * 2);
  float* hn1   = (float*)alloc(512 * 4);
  float* hn2   = (float*)alloc(512 * 4);
  float* hnE   = (float*)alloc(256 * 4);
  float* w1q   = (float*)alloc(256 * 9 * 4);
  float* w2q   = (float*)alloc(256 * 9 * 4);
  float* h1    = (float*)alloc((size_t)64 * HW * 4);
  float* h2    = (float*)alloc((size_t)1024 * HW * 4);
  float* h3    = (float*)alloc((size_t)1024 * HW * 4);
  int*   idx1  = (int*)alloc(64 * 4);
  int*   idx2  = (int*)alloc(1024 * 4);
  int*   cnts  = (int*)alloc((size_t)64 * 256 * 4);
  float* opre  = (float*)alloc((size_t)64 * HW * 4);
  float* stats = (float*)alloc(32 * 4);

  // prep: bf16 codebooks + half-norms; quantize filters; zero counts
  k_prep_codes<<<512, 256, 0, stream>>>(emb_f1, ef1b, hn1, HW);
  k_prep_codes<<<512, 256, 0, stream>>>(emb_f2, ef2b, hn2, HW);
  k_prep_codes<<<256, 256, 0, stream>>>(emb_end, eendb, hnE, HW);
  k_vq_w<<<1, 256, 0, stream>>>(pw_w1, emb_pw1, w1q);
  k_vq_w<<<1, 256, 0, stream>>>(pw_w2, emb_pw2, w2q);
  k_zero_counts<<<64, 256, 0, stream>>>(cnts);

  // stage 1: vq(x) vs emb_f1 (WMMA), gather
  k_vq_rows<<<4, 512, 0, stream>>>(x, ef1b, hn1, idx1);
  k_gather<<<64, 256, 0, stream>>>(idx1, emb_f1, h1, HW);

  // stage 2: depthwise conv 1
  k_dwconv1<<<1024, 256, 0, stream>>>(h1, w1q, h2);

  // stage 3: vq(h2) vs emb_f2 (WMMA), gather
  k_vq_rows<<<64, 512, 0, stream>>>(h2, ef2b, hn2, idx2);
  k_gather<<<1024, 256, 0, stream>>>(idx2, emb_f2, h3, HW);

  // stage 4+5 fused: conv2 on the fly + vq(emb_end) via WMMA + counting
  k_conv2_vqend<<<256, 512, 0, stream>>>(h3, w2q, eendb, hnE, cnts);

  // stage 6: counts @ emb_end, BN stats, final BN + residual + ReLU
  k_outpre<<<64, 256, 0, stream>>>(emb_end, cnts, opre);
  k_bnstats<<<OUPC, 256, 0, stream>>>(opre, stats);
  k_final<<<(BATCH * OUPC * HW + 255) / 256, 256, 0, stream>>>(opre, x, stats, gamma, beta, out);

  (void)in_sizes; (void)n_in; (void)out_size; (void)ws_size;
}
